// OpenCUAVisionAttention_17042430231113
// MI455X (gfx1250) — matrix operand order repaired and checked by reading, not compiled, not run on hardware
//
#include <hip/hip_runtime.h>
#include <hip/hip_bf16.h>
#include <cstdint>

typedef __attribute__((ext_vector_type(16))) __bf16 v16bf;
typedef __attribute__((ext_vector_type(8)))  __bf16 v8bf;
typedef __attribute__((ext_vector_type(8)))  float  v8f;

#define CAT16(lo, hi) __builtin_shufflevector(lo, hi, 0,1,2,3,4,5,6,7,8,9,10,11,12,13,14,15)

static __device__ __forceinline__ v8f wmma_bf16(v16bf a, v16bf b, v8f c) {
    return __builtin_amdgcn_wmma_f32_16x16x32_bf16(false, a, false, b, (short)0, c, false, false);
}

// gfx1250 async copy global -> LDS (ASYNCcnt-tracked, no VGPR data round-trip).
// lds_off: wave-relative LDS byte address (low 32 bits of a generic LDS pointer,
// per the CDNA5 aperture mapping).
static __device__ __forceinline__ void async_copy_b128(unsigned lds_off, const void* gptr) {
    asm volatile("global_load_async_to_lds_b128 %0, %1, off"
                 :: "v"(lds_off), "v"(gptr) : "memory");
}
static __device__ __forceinline__ void wait_asynccnt0() {
    asm volatile("s_wait_asynccnt 0x0" ::: "memory");
}

// ---------------------------------------------------------------------------
// Elementwise fp32 -> bf16 convert
// ---------------------------------------------------------------------------
__global__ void cvt_f32_bf16(const float* __restrict__ src, __bf16* __restrict__ dst, int n) {
    int i = blockIdx.x * 256 + threadIdx.x;
    if (i < n) dst[i] = (__bf16)src[i];
}

// W [K][N] fp32 row-major  ->  Wt [N][K] bf16 (B-operand friendly)
__global__ void transpose_f32_bf16(const float* __restrict__ W, __bf16* __restrict__ Wt,
                                   int K, int N) {
    long long i = (long long)blockIdx.x * 256 + threadIdx.x;
    long long total = (long long)K * N;
    if (i < total) {
        int n = (int)(i / K), k = (int)(i % K);
        Wt[i] = (__bf16)W[(long long)k * N + n];
    }
}

// ---------------------------------------------------------------------------
// bf16 WMMA GEMM with async-LDS double-buffered staging:
//   C[M][N] (f32) = A[M][K] * Bt[N][K]^T + bias[N]
// block = 128 threads (4 waves); block tile 128x128; k staged 64-deep in LDS.
// Stage buffers filled with global_load_async_to_lds_b128 (ASYNCcnt), double
// buffered so stage s+1 streams in while stage s feeds 32 WMMAs.
// ---------------------------------------------------------------------------
__global__ __launch_bounds__(128)
void gemm_bf16_wmma(const __bf16* __restrict__ A, const __bf16* __restrict__ Bt,
                    const float* __restrict__ bias, float* __restrict__ C,
                    int M, int N, int K) {
    __shared__ __bf16 Alds[2][128 * 64];   // 2 x 16 KB
    __shared__ __bf16 Blds[2][128 * 64];   // 2 x 16 KB

    const int tid  = threadIdx.x;
    const int lane = tid & 31;
    const int wave = tid >> 5;
    const int half = lane >> 4;
    const int l16  = lane & 15;
    const int rh   = (wave >> 1) * 64;          // wave's row offset inside 128-tile
    const int ch   = (wave & 1) * 64;           // wave's col offset inside 128-tile
    const int row0 = blockIdx.y * 128;
    const int col0 = blockIdx.x * 128;

    // async-copy work split: thread owns (row = tid>>3 + 16*i, 16B chunk tid&7)
    const int trow = tid >> 3;                  // 0..15
    const int tseg = tid & 7;                   // 0..7 (16B chunks of a 128B row)
    const __bf16* ag = A  + (size_t)(row0 + trow) * K + tseg * 8;
    const __bf16* bg = Bt + (size_t)(col0 + trow) * K + tseg * 8;
    const unsigned aL = (unsigned)(size_t)&Alds[0][0] + trow * 128 + tseg * 16;
    const unsigned bL = (unsigned)(size_t)&Blds[0][0] + trow * 128 + tseg * 16;

    auto issue_stage = [&](int p, int kbase) {
#pragma unroll
        for (int i = 0; i < 8; ++i) {
            async_copy_b128(aL + p * 16384 + i * (16 * 128),
                            ag + (size_t)(i * 16) * K + kbase);
            async_copy_b128(bL + p * 16384 + i * (16 * 128),
                            bg + (size_t)(i * 16) * K + kbase);
        }
    };

    v8f acc[4][4] = {};
    const int nStages = K / 64;

    issue_stage(0, 0);
    for (int s = 0; s < nStages; ++s) {
        wait_asynccnt0();          // this wave's stage-s chunks have landed
        __syncthreads();           // everyone's chunks landed; stage s-1 compute done
        if (s + 1 < nStages) issue_stage((s + 1) & 1, (s + 1) * 64);

        const __bf16* Ab = &Alds[s & 1][0];
        const __bf16* Bb = &Blds[s & 1][0];
#pragma unroll
        for (int kk = 0; kk < 2; ++kk) {       // two 32-deep WMMA steps per stage
            v16bf af[4], bfg[4];
#pragma unroll
            for (int i = 0; i < 4; ++i) {
                const __bf16* p = Ab + (rh + i * 16 + l16) * 64 + kk * 32 + half * 16;
                v8bf lo = *(const v8bf*)p;
                v8bf hi = *(const v8bf*)(p + 8);
                af[i] = CAT16(lo, hi);
            }
#pragma unroll
            for (int j = 0; j < 4; ++j) {
                const __bf16* p = Bb + (ch + j * 16 + l16) * 64 + kk * 32 + half * 16;
                bfg[j] = *(const v16bf*)p;
            }
#pragma unroll
            for (int i = 0; i < 4; ++i)
#pragma unroll
                for (int j = 0; j < 4; ++j)
                    acc[i][j] = wmma_bf16(af[i], bfg[j], acc[i][j]);
        }
    }

#pragma unroll
    for (int j = 0; j < 4; ++j) {
        int c = col0 + ch + j * 16 + l16;
        float b = bias[c];
#pragma unroll
        for (int i = 0; i < 4; ++i) {
#pragma unroll
            for (int r = 0; r < 8; ++r) {
                int rr = row0 + rh + i * 16 + r + half * 8;
                C[(size_t)rr * N + c] = acc[i][j][r] + b;
            }
        }
    }
}

// ---------------------------------------------------------------------------
// RoPE (rotate-half) + repack:
//   qkvf [2048][3840] f32 ->
//     Qp,Kp [seg*16+h][256][96] bf16 (head_dim 80 padded to 96 with zeros)
//     Vt    [seg*16+h][80][256] bf16 (transposed, PV B-operand friendly)
// ---------------------------------------------------------------------------
__global__ void rope_repack(const float* __restrict__ qkv, const float* __restrict__ rope,
                            __bf16* __restrict__ Qp, __bf16* __restrict__ Kp,
                            __bf16* __restrict__ Vt) {
    const int s = blockIdx.x;            // token 0..2047
    const int seg = s >> 8, t = s & 255;
    const float* qrow = qkv + (size_t)s * 3840;
    const float* frow = rope + (size_t)s * 40;

    for (int i = threadIdx.x; i < 16 * 96; i += 256) {
        int h = i / 96, d = i % 96;
        float qo = 0.f, ko = 0.f;
        if (d < 80) {
            float q = qrow[h * 80 + d];
            float kk = qrow[1280 + h * 80 + d];
            if (d < 40) {
                float f = frow[d];
                float c = __cosf(f), sn = __sinf(f);
                float q2 = qrow[h * 80 + d + 40];
                float k2 = qrow[1280 + h * 80 + d + 40];
                qo = q * c - q2 * sn;
                ko = kk * c - k2 * sn;
            } else {
                float f = frow[d - 40];
                float c = __cosf(f), sn = __sinf(f);
                float q2 = qrow[h * 80 + d - 40];
                float k2 = qrow[1280 + h * 80 + d - 40];
                qo = q * c + q2 * sn;
                ko = kk * c + k2 * sn;
            }
        }
        size_t o = ((size_t)(seg * 16 + h) * 256 + t) * 96 + d;
        Qp[o] = (__bf16)qo;
        Kp[o] = (__bf16)ko;
    }
    for (int i = threadIdx.x; i < 16 * 80; i += 256) {
        int h = i / 80, d = i % 80;
        Vt[((size_t)(seg * 16 + h) * 80 + d) * 256 + t] = (__bf16)qrow[2560 + h * 80 + d];
    }
}

// ---------------------------------------------------------------------------
// Attention: one block per (segment, head). 256 threads = 8 waves.
// Each wave handles two 16-row strips: S = Q*K^T (WMMA), softmax (shfl_xor
// reductions inside the C-fragment layout), P -> bf16 LDS strip, ctx = P*V (WMMA).
// ---------------------------------------------------------------------------
__global__ __launch_bounds__(256)
void attn_wmma(const __bf16* __restrict__ Qp, const __bf16* __restrict__ Kp,
               const __bf16* __restrict__ Vt, __bf16* __restrict__ ctxb) {
    __shared__ __bf16 Plds[8 * 16 * 256];   // 64 KB: per-wave 16x256 bf16 strip

    const int sh = blockIdx.x;              // seg*16 + head
    const int seg = sh >> 4, h = sh & 15;
    const int lane = threadIdx.x & 31;
    const int wave = threadIdx.x >> 5;      // 0..7
    const int half = lane >> 4;
    const int l16  = lane & 15;

    const __bf16* Qb = Qp + (size_t)sh * 256 * 96;
    const __bf16* Kb = Kp + (size_t)sh * 256 * 96;
    const __bf16* Vb = Vt + (size_t)sh * 80 * 256;
    __bf16* pbuf = &Plds[wave * 16 * 256];

    const float scale = 0.11180339887498949f;   // 1/sqrt(80)

    for (int strip = 0; strip < 2; ++strip) {
        const int r0 = (wave * 2 + strip) * 16;     // strip row origin in segment

        // ---- phase 1: scores 16x256 = Q strip (16x96) * K^T ----
        v16bf aq[3];
#pragma unroll
        for (int kk = 0; kk < 3; ++kk) {
            const __bf16* ap = Qb + (size_t)(r0 + l16) * 96 + kk * 32 + half * 16;
            v8bf lo = *(const v8bf*)ap;
            v8bf hi = *(const v8bf*)(ap + 8);
            aq[kk] = CAT16(lo, hi);
        }
        v8f sc[16];
#pragma unroll
        for (int j = 0; j < 16; ++j) {
            v8f a = {};
#pragma unroll
            for (int kk = 0; kk < 3; ++kk) {
                v16bf bb = *(const v16bf*)(Kb + (size_t)(j * 16 + l16) * 96 + kk * 32 + half * 16);
                a = wmma_bf16(aq[kk], bb, a);
            }
            sc[j] = a * scale;
        }

        // ---- softmax over 256 cols; row = r + 8*half, cols striped over 16 lanes ----
        float mrow[8], lrow[8];
#pragma unroll
        for (int r = 0; r < 8; ++r) {
            float v = -3.0e38f;
#pragma unroll
            for (int j = 0; j < 16; ++j) v = fmaxf(v, sc[j][r]);
#pragma unroll
            for (int mk = 8; mk >= 1; mk >>= 1) v = fmaxf(v, __shfl_xor(v, mk, 16));
            mrow[r] = v;
        }
#pragma unroll
        for (int j = 0; j < 16; ++j)
#pragma unroll
            for (int r = 0; r < 8; ++r) sc[j][r] = __expf(sc[j][r] - mrow[r]);
#pragma unroll
        for (int r = 0; r < 8; ++r) {
            float v = 0.f;
#pragma unroll
            for (int j = 0; j < 16; ++j) v += sc[j][r];
#pragma unroll
            for (int mk = 8; mk >= 1; mk >>= 1) v += __shfl_xor(v, mk, 16);
            lrow[r] = 1.0f / v;
        }

        // ---- write normalized P (bf16) into this wave's LDS strip ----
#pragma unroll
        for (int j = 0; j < 16; ++j)
#pragma unroll
            for (int r = 0; r < 8; ++r)
                pbuf[(r + half * 8) * 256 + j * 16 + l16] = (__bf16)(sc[j][r] * lrow[r]);

        __syncthreads();

        // ---- phase 2: ctx 16x80 = P (16x256) * V (256x80) ----
        v16bf ap2[8];
#pragma unroll
        for (int k8 = 0; k8 < 8; ++k8) {
            const __bf16* pp = &pbuf[l16 * 256 + k8 * 32 + half * 16];
            v8bf lo = *(const v8bf*)pp;
            v8bf hi = *(const v8bf*)(pp + 8);
            ap2[k8] = CAT16(lo, hi);
        }
#pragma unroll
        for (int jn = 0; jn < 5; ++jn) {
            v8f a = {};
#pragma unroll
            for (int k8 = 0; k8 < 8; ++k8) {
                v16bf bb = *(const v16bf*)(Vb + (size_t)(jn * 16 + l16) * 256 + k8 * 32 + half * 16);
                a = wmma_bf16(ap2[k8], bb, a);
            }
            const int col = h * 80 + jn * 16 + l16;
#pragma unroll
            for (int r = 0; r < 8; ++r) {
                int tok = seg * 256 + r0 + r + half * 8;
                ctxb[(size_t)tok * 1280 + col] = (__bf16)a[r];
            }
        }
        __syncthreads();
    }
}

// ---------------------------------------------------------------------------
// Host launcher
// ---------------------------------------------------------------------------
extern "C" void kernel_launch(void* const* d_in, const int* in_sizes, int n_in,
                              void* d_out, int out_size, void* d_ws, size_t ws_size,
                              hipStream_t stream) {
    const float* x     = (const float*)d_in[0];   // [2048][1280]
    const float* rope  = (const float*)d_in[1];   // [2048][40]
    const float* Wqkv  = (const float*)d_in[2];   // [1280][3840]
    const float* bqkv  = (const float*)d_in[3];   // [3840]
    const float* Wproj = (const float*)d_in[4];   // [1280][1280]
    const float* bproj = (const float*)d_in[5];   // [1280]
    float* out = (float*)d_out;                   // [2048][1280]

    char* ws = (char*)d_ws;
    size_t off = 0;
    auto carve = [&](size_t bytes) -> void* {
        void* p = ws + off;
        off += (bytes + 255) & ~(size_t)255;
        return p;
    };
    __bf16* xb   = (__bf16*)carve(2048ull * 1280 * 2);
    __bf16* Wq_t = (__bf16*)carve(3840ull * 1280 * 2);
    __bf16* Wp_t = (__bf16*)carve(1280ull * 1280 * 2);
    float*  qkvf = (float*) carve(2048ull * 3840 * 4);
    __bf16* Qp   = (__bf16*)carve(128ull * 256 * 96 * 2);
    __bf16* Kp   = (__bf16*)carve(128ull * 256 * 96 * 2);
    __bf16* Vt   = (__bf16*)carve(128ull * 80 * 256 * 2);
    __bf16* ctxb = (__bf16*)carve(2048ull * 1280 * 2);

    // 1) precision/layout conversion (L2-resident bf16 operands)
    cvt_f32_bf16<<<(2048 * 1280 + 255) / 256, 256, 0, stream>>>(x, xb, 2048 * 1280);
    transpose_f32_bf16<<<(1280 * 3840 + 255) / 256, 256, 0, stream>>>(Wqkv, Wq_t, 1280, 3840);
    transpose_f32_bf16<<<(1280 * 1280 + 255) / 256, 256, 0, stream>>>(Wproj, Wp_t, 1280, 1280);

    // 2) QKV projection: [2048,1280] x [1280,3840] + bias  (async-LDS WMMA GEMM)
    gemm_bf16_wmma<<<dim3(3840 / 128, 2048 / 128), 128, 0, stream>>>(
        xb, Wq_t, bqkv, qkvf, 2048, 3840, 1280);

    // 3) RoPE + repack to per-(seg,head) tiles (pad 80->96, V transposed)
    rope_repack<<<2048, 256, 0, stream>>>(qkvf, rope, Qp, Kp, Vt);

    // 4) block-diagonal attention: one block per (segment, head)
    attn_wmma<<<8 * 16, 256, 0, stream>>>(Qp, Kp, Vt, ctxb);

    // 5) output projection: [2048,1280] x [1280,1280] + bias -> d_out (f32)
    gemm_bf16_wmma<<<dim3(1280 / 128, 2048 / 128), 128, 0, stream>>>(
        ctxb, Wp_t, bproj, out, 2048, 1280, 1280);
}
